// CachedMistralAttention_72438918414684
// MI455X (gfx1250) — compile-verified, matrix operand order checked by reading, not run on hardware
//
#include <hip/hip_runtime.h>
#include <stdint.h>

typedef __attribute__((ext_vector_type(16))) __bf16 v16bf;
typedef __attribute__((ext_vector_type(8)))  float  v8f;

union Frag { v16bf v; uint4 q[2]; };

__device__ __forceinline__ uint16_t f2bf(float f) {
  union { float f; uint32_t u; } c; c.f = f;
  uint32_t u = c.u;
  return (uint16_t)((u + 0x7FFFu + ((u >> 16) & 1u)) >> 16);
}
__device__ __forceinline__ float bf2f(uint16_t h) {
  union { uint32_t u; float f; } c; c.u = ((uint32_t)h) << 16;
  return c.f;
}
__device__ __forceinline__ v8f wmma_bf16(v16bf a, v16bf b, v8f c) {
  // D = A(16x32 bf16) * B(32x16 bf16) + C(16x16 f32)
  return __builtin_amdgcn_wmma_f32_16x16x32_bf16(false, a, false, b, (short)0, c, false, false);
}

// ---------------------------------------------------------------------------
// Tiled GEMM: C[M,N] = A[M,K] * B[K,N].  A fp32 (converted) or bf16; B fp32
// weights converted to bf16 while staging; C stored bf16 or fp32.
// Block = 256 threads = 8 waves; tile 128x128, K-step 32.
// Wave (wm 0..3, wn 0..1) owns a 32x64 sub-tile = 2x4 WMMA accumulators.
// LDS layout: row-major with stride 40 bf16 so every WMMA fragment is two
// aligned 16B reads:  elems 0..7 = K kb..kb+7, elems 8..15 = K kb+16..kb+23,
// kb = (lane<16) ? 0 : 8   (per CDNA5 ISA 16-bit fragment layout).
// ---------------------------------------------------------------------------
template<bool ABF16, bool CBF16>
__global__ void __launch_bounds__(256)
gemm_wmma(const void* __restrict__ Aptr, const float* __restrict__ B,
          void* __restrict__ Cptr, int K, int lda, int ldb, int ldc) {
  __shared__ __align__(16) uint16_t lA[128 * 40];
  __shared__ __align__(16) uint16_t lB[128 * 40];
  const int tid   = threadIdx.x;
  const int lane  = tid & 31;
  const int wave  = tid >> 5;
  const int wm    = wave & 3;
  const int wn    = wave >> 2;
  const int m0    = blockIdx.y * 128;
  const int n0    = blockIdx.x * 128;
  const int lrow  = lane & 15;
  const int half  = lane >> 4;
  const int khalf = half * 8;

  v8f acc[2][4];
#pragma unroll
  for (int i = 0; i < 2; ++i)
#pragma unroll
    for (int j = 0; j < 4; ++j) acc[i][j] = {};

  for (int k0 = 0; k0 < K; k0 += 32) {
    if constexpr (ABF16) {
      const uint16_t* A = (const uint16_t*)Aptr;
#pragma unroll
      for (int i = 0; i < 2; ++i) {
        int idx = tid + i * 256;          // 128 rows x 4 uint4
        int r = idx >> 2, c = idx & 3;
        uint4 v = *(const uint4*)(A + (size_t)(m0 + r) * lda + k0 + c * 8);
        *(uint4*)&lA[r * 40 + c * 8] = v;
      }
    } else {
      const float* A = (const float*)Aptr;
#pragma unroll
      for (int i = 0; i < 4; ++i) {
        int idx = idx = tid + i * 256;    // 128 rows x 8 float4
        int r = idx >> 3, c = idx & 7;
        float4 v = *(const float4*)(A + (size_t)(m0 + r) * lda + k0 + c * 4);
        uint2 p;
        p.x = (uint32_t)f2bf(v.x) | ((uint32_t)f2bf(v.y) << 16);
        p.y = (uint32_t)f2bf(v.z) | ((uint32_t)f2bf(v.w) << 16);
        *(uint2*)&lA[r * 40 + c * 4] = p;
      }
    }
    // B tile 32x128, transposed into lB[n][k]
#pragma unroll
    for (int i = 0; i < 4; ++i) {
      int idx = tid + i * 256;            // 32 rows x 32 float4
      int r = idx >> 5, c = idx & 31;
      float4 v = *(const float4*)(B + (size_t)(k0 + r) * ldb + n0 + c * 4);
      int nb = c * 4;
      lB[(nb + 0) * 40 + r] = f2bf(v.x);
      lB[(nb + 1) * 40 + r] = f2bf(v.y);
      lB[(nb + 2) * 40 + r] = f2bf(v.z);
      lB[(nb + 3) * 40 + r] = f2bf(v.w);
    }
    __syncthreads();

    v16bf afrag[2], bfrag[4];
#pragma unroll
    for (int mi = 0; mi < 2; ++mi) {
      Frag f;
      const uint16_t* p = &lA[(wm * 32 + mi * 16 + lrow) * 40 + khalf];
      f.q[0] = *(const uint4*)p;
      f.q[1] = *(const uint4*)(p + 16);
      afrag[mi] = f.v;
    }
#pragma unroll
    for (int ni = 0; ni < 4; ++ni) {
      Frag f;
      const uint16_t* p = &lB[(wn * 64 + ni * 16 + lrow) * 40 + khalf];
      f.q[0] = *(const uint4*)p;
      f.q[1] = *(const uint4*)(p + 16);
      bfrag[ni] = f.v;
    }
#pragma unroll
    for (int mi = 0; mi < 2; ++mi)
#pragma unroll
      for (int ni = 0; ni < 4; ++ni)
        acc[mi][ni] = wmma_bf16(afrag[mi], bfrag[ni], acc[mi][ni]);
    __syncthreads();
  }

  // C layout: vgpr r -> row (r + 8*half), col = lane&15
#pragma unroll
  for (int mi = 0; mi < 2; ++mi)
#pragma unroll
    for (int ni = 0; ni < 4; ++ni) {
      int nc = n0 + wn * 64 + ni * 16 + lrow;
#pragma unroll
      for (int r = 0; r < 8; ++r) {
        int mr = m0 + wm * 32 + mi * 16 + r + half * 8;
        float v = acc[mi][ni][r];
        if constexpr (CBF16) ((uint16_t*)Cptr)[(size_t)mr * ldc + nc] = f2bf(v);
        else                 ((float*)Cptr)[(size_t)mr * ldc + nc] = v;
      }
    }
}

// ---------------------------------------------------------------------------
// RoPE (rotate-half), in place on bf16 Q [2048,32,128] and K [2048,8,128].
// ---------------------------------------------------------------------------
__global__ void __launch_bounds__(256)
rope_kernel(uint16_t* __restrict__ Q, uint16_t* __restrict__ Kb) {
  const int QP = 2048 * 32 * 64;
  const int KP = 2048 * 8 * 64;
  int idx = blockIdx.x * 256 + threadIdx.x;
  if (idx >= QP + KP) return;
  uint16_t* base;
  int pos, j;
  if (idx < QP) {
    pos = idx >> 11;                // / (32*64)
    int rem = idx & 2047;
    int hd = rem >> 6; j = rem & 63;
    base = Q + (size_t)pos * 4096 + hd * 128;
  } else {
    int t = idx - QP;
    pos = t >> 9;                   // / (8*64)
    int rem = t & 511;
    int hd = rem >> 6; j = rem & 63;
    base = Kb + (size_t)pos * 1024 + hd * 128;
  }
  // inv_freq = 10000^(-j/64) = exp(-j * ln(10000)/64)
  float inv = expf(-0.14391156831212788f * (float)j);
  float f = (float)pos * inv;
  float sn, cs;
  sincosf(f, &sn, &cs);
  float x1 = bf2f(base[j]);
  float x2 = bf2f(base[j + 64]);
  base[j]      = f2bf(x1 * cs - x2 * sn);
  base[j + 64] = f2bf(x2 * cs + x1 * sn);
}

// ---------------------------------------------------------------------------
// Flash attention (causal, GQA x4). One block per (q-block of 128, head).
// 8 waves, each owns 16 q-rows. K-blocks of 64 keys staged to LDS:
//   lK[key][h]  (stride 136) -> B-fragments for S = Q*K^T
//   lV[h][key]  (stride 72)  -> B-fragments for O += P*V
//   lP[wave][m][kk] (stride 72): C-layout -> A-fragment reshape for P
// ---------------------------------------------------------------------------
__global__ void __launch_bounds__(256)
attn_kernel(const uint16_t* __restrict__ Q, const uint16_t* __restrict__ Kg,
            const uint16_t* __restrict__ Vg, uint16_t* __restrict__ Og) {
  __shared__ __align__(16) uint16_t lK[64 * 136];
  __shared__ __align__(16) uint16_t lV[128 * 72];
  __shared__ __align__(16) uint16_t lP[8 * 16 * 72];
  const int tid    = threadIdx.x;
  const int lane   = tid & 31;
  const int wave   = tid >> 5;
  const int qblock = blockIdx.x;
  const int head   = blockIdx.y;
  const int kvoff  = (head >> 2) * 128;
  const int lrow   = lane & 15;
  const int half   = lane >> 4;
  const int khalf  = half * 8;
  const int qrow   = qblock * 128 + wave * 16;  // wave's first q row

  // Q fragments over head_dim (4 chunks of 32)
  v16bf qfrag[4];
  {
    const uint16_t* qp = Q + (size_t)(qrow + lrow) * 4096 + head * 128;
#pragma unroll
    for (int f = 0; f < 4; ++f) {
      Frag fr;
      fr.q[0] = *(const uint4*)(qp + f * 32 + khalf);
      fr.q[1] = *(const uint4*)(qp + f * 32 + khalf + 16);
      qfrag[f] = fr.v;
    }
  }

  v8f o[8];
#pragma unroll
  for (int i = 0; i < 8; ++i) o[i] = {};
  float rmax[8], rsum[8];
#pragma unroll
  for (int r = 0; r < 8; ++r) { rmax[r] = -3e38f; rsum[r] = 0.0f; }

  const float scale = 0.08838834764831845f;  // 1/sqrt(128)
  const int nkb = (qblock + 1) * 2;
  uint16_t* lPw = lP + wave * 16 * 72;

  for (int kbi = 0; kbi < nkb; ++kbi) {
    const int kb0 = kbi * 64;
    // stage K (row-major) and V (transposed): 64 keys x 128 h
#pragma unroll
    for (int i = 0; i < 4; ++i) {
      int idx = tid + i * 256;
      int r = idx >> 4, c = idx & 15;
      uint4 kv = *(const uint4*)(Kg + (size_t)(kb0 + r) * 1024 + kvoff + c * 8);
      *(uint4*)&lK[r * 136 + c * 8] = kv;
      union { uint4 u; uint16_t s[8]; } vv;
      vv.u = *(const uint4*)(Vg + (size_t)(kb0 + r) * 1024 + kvoff + c * 8);
#pragma unroll
      for (int e = 0; e < 8; ++e) lV[(c * 8 + e) * 72 + r] = vv.s[e];
    }
    __syncthreads();

    // scores: 4 key-subtiles of 16, each accumulated over 4 h-chunks
    v8f s[4];
#pragma unroll
    for (int ks = 0; ks < 4; ++ks) {
      v8f a = {};
#pragma unroll
      for (int f = 0; f < 4; ++f) {
        Frag bf;
        const uint16_t* kp = &lK[(ks * 16 + lrow) * 136 + f * 32 + khalf];
        bf.q[0] = *(const uint4*)kp;
        bf.q[1] = *(const uint4*)(kp + 16);
        a = wmma_bf16(qfrag[f], bf.v, a);
      }
      s[ks] = a;
    }

    // online softmax update
#pragma unroll
    for (int r = 0; r < 8; ++r) {
      int mg = qrow + r + half * 8;
      float mx = -3e38f;
#pragma unroll
      for (int ks = 0; ks < 4; ++ks) {
        int kg = kb0 + ks * 16 + lrow;
        float v = s[ks][r] * scale;
        v = (kg <= mg) ? v : -3e38f;     // causal mask (tril)
        s[ks][r] = v;
        mx = fmaxf(mx, v);
      }
#pragma unroll
      for (int d = 1; d < 16; d <<= 1) mx = fmaxf(mx, __shfl_xor(mx, d, 32));
      float nm = fmaxf(rmax[r], mx);
      float sc = __expf(rmax[r] - nm);
      rmax[r] = nm;
      rsum[r] *= sc;
#pragma unroll
      for (int ht = 0; ht < 8; ++ht) o[ht][r] *= sc;
      float rs = 0.0f;
#pragma unroll
      for (int ks = 0; ks < 4; ++ks) {
        float p = __expf(s[ks][r] - nm);
        s[ks][r] = p;
        rs += p;
      }
#pragma unroll
      for (int d = 1; d < 16; d <<= 1) rs += __shfl_xor(rs, d, 32);
      rsum[r] += rs;
    }

    // reshape P: C-layout regs -> per-wave LDS -> A-fragments
#pragma unroll
    for (int ks = 0; ks < 4; ++ks)
#pragma unroll
      for (int r = 0; r < 8; ++r)
        lPw[(r + half * 8) * 72 + ks * 16 + lrow] = f2bf(s[ks][r]);

    v16bf pfrag[2];
#pragma unroll
    for (int c = 0; c < 2; ++c) {
      Frag fr;
      const uint16_t* pp = &lPw[lrow * 72 + c * 32 + khalf];
      fr.q[0] = *(const uint4*)pp;
      fr.q[1] = *(const uint4*)(pp + 16);
      pfrag[c] = fr.v;
    }
    // O += P * V  (8 h-tiles x 2 kk-chunks)
#pragma unroll
    for (int ht = 0; ht < 8; ++ht)
#pragma unroll
      for (int c = 0; c < 2; ++c) {
        Frag bf;
        const uint16_t* vp = &lV[(ht * 16 + lrow) * 72 + c * 32 + khalf];
        bf.q[0] = *(const uint4*)vp;
        bf.q[1] = *(const uint4*)(vp + 16);
        o[ht] = wmma_bf16(pfrag[c], bf.v, o[ht]);
      }
    __syncthreads();
  }

  // normalize and store attn output [q, head*128 + h] as bf16
#pragma unroll
  for (int r = 0; r < 8; ++r) {
    float inv = 1.0f / rsum[r];
    int mg = qrow + r + half * 8;
    uint16_t* op = Og + (size_t)mg * 4096 + head * 128;
#pragma unroll
    for (int ht = 0; ht < 8; ++ht)
      op[ht * 16 + lrow] = f2bf(o[ht][r] * inv);
  }
}

// ---------------------------------------------------------------------------
extern "C" void kernel_launch(void* const* d_in, const int* in_sizes, int n_in,
                              void* d_out, int out_size, void* d_ws, size_t ws_size,
                              hipStream_t stream) {
  (void)in_sizes; (void)n_in; (void)out_size; (void)ws_size;
  const float* hs = (const float*)d_in[0];
  // d_in[1] is the causal tril mask -> applied analytically in attn_kernel
  const float* wq = (const float*)d_in[2];
  const float* wk = (const float*)d_in[3];
  const float* wv = (const float*)d_in[4];
  const float* wo = (const float*)d_in[5];
  float* out = (float*)d_out;

  uint16_t* qb   = (uint16_t*)d_ws;                 // 2048*4096 bf16
  uint16_t* kb   = qb + (size_t)2048 * 4096;        // 2048*1024 bf16
  uint16_t* vb   = kb + (size_t)2048 * 1024;        // 2048*1024 bf16
  uint16_t* attn = vb + (size_t)2048 * 1024;        // 2048*4096 bf16

  dim3 blk(256);
  // QKV projections (fp32 in, bf16 out, fp32 accum in WMMA)
  gemm_wmma<false, true><<<dim3(32, 16), blk, 0, stream>>>(hs, wq, qb, 4096, 4096, 4096, 4096);
  gemm_wmma<false, true><<<dim3(8, 16),  blk, 0, stream>>>(hs, wk, kb, 4096, 4096, 1024, 1024);
  gemm_wmma<false, true><<<dim3(8, 16),  blk, 0, stream>>>(hs, wv, vb, 4096, 4096, 1024, 1024);
  // RoPE on Q and K
  rope_kernel<<<20480, blk, 0, stream>>>(qb, kb);
  // causal flash attention, GQA
  attn_kernel<<<dim3(16, 32), blk, 0, stream>>>(qb, kb, vb, attn);
  // output projection (bf16 A, fp32 out)
  gemm_wmma<true, false><<<dim3(32, 16), blk, 0, stream>>>(attn, wo, out, 4096, 4096, 4096, 4096);
}
